// OCAB_10161892622410
// MI455X (gfx1250) — matrix operand order repaired and checked by reading, not compiled, not run on hardware
//
#include <hip/hip_runtime.h>
#include <hip/hip_bf16.h>
#include <math.h>

typedef _Float16 v16h __attribute__((ext_vector_type(16)));
typedef _Float16 v8h  __attribute__((ext_vector_type(8)));
typedef float    v8f  __attribute__((ext_vector_type(8)));

#define C_    192
#define NH_   6
#define D_    32
#define WSZ   16
#define OWSZ  24
#define NKT   576           // 24*24 keys per window
#define NTOK  32768         // 2*128*128 tokens
#define KT_LD 584           // padded LDS row for K^T / scores
#define SC_LD 584

// ---- WMMA fragment loaders (CDNA5 wave32 layouts) ----
// A (16x32 f16): lane<16 holds row=lane&15, K {0..7,16..23}; lane>=16 K {8..15,24..31}
__device__ __forceinline__ v16h ldA(const _Float16* p, int hs) {
  v16h a;
  ((v8h*)&a)[0] = *(const v8h*)(p + hs * 8);
  ((v8h*)&a)[1] = *(const v8h*)(p + 16 + hs * 8);
  return a;
}
// B (32x16 f16): lane holds column=lane&15, K {0..7,16..23} (lane<16) / {8..15,24..31} (lane>=16)
__device__ __forceinline__ v16h ldB(const _Float16* p, int ld) {
  v16h b;
  const int ln = threadIdx.x & 31, bh = ln >> 4, bn = ln & 15;
#pragma unroll
  for (int s = 0; s < 16; ++s) {
    int k = (s < 8) ? (bh * 8 + s) : (16 + bh * 8 + (s - 8));
    b[s] = p[(size_t)k * ld + bn];
  }
  return b;
}

// ---- weight prep: f32 -> f16, slice/transpose as needed ----
__global__ __launch_bounds__(256) void prep_w_k(
    const float* __restrict__ qkv_w, const float* __restrict__ dproj_w,
    const float* __restrict__ proj_w, const float* __restrict__ fc1_w,
    const float* __restrict__ fc2_w,
    _Float16* __restrict__ wkv, _Float16* __restrict__ dT,
    _Float16* __restrict__ pw, _Float16* __restrict__ f1,
    _Float16* __restrict__ f2) {
  int i = blockIdx.x * 256 + threadIdx.x;
  if (i < 73728) { int k = i / 384, n = i % 384; wkv[i] = (_Float16)qkv_w[k * 576 + 192 + n]; return; }
  i -= 73728;
  if (i < 36864) { int k = i / 192, n = i % 192; dT[i] = (_Float16)dproj_w[n * 192 + k]; return; }
  i -= 36864;
  if (i < 36864) { pw[i] = (_Float16)proj_w[i]; return; }
  i -= 36864;
  if (i < 73728) { f1[i] = (_Float16)fc1_w[i]; return; }
  i -= 73728;
  if (i < 73728) { f2[i] = (_Float16)fc2_w[i]; return; }
}

// ---- rel-pos bias gather: bb[h][q][k] = tbl[rpi[q*576+k]*6 + h] ----
__global__ __launch_bounds__(256) void bias_k(const int* __restrict__ rpi,
                                              const float* __restrict__ tbl,
                                              float* __restrict__ bb) {
  int i = blockIdx.x * 256 + threadIdx.x;
  if (i >= NH_ * 256 * NKT) return;
  int h = i / (256 * NKT);
  int rem = i - h * (256 * NKT);   // q*576 + k
  bb[i] = tbl[rpi[rem] * NH_ + h];
}

// ---- LN (optional) + layout pack to token-major f16 ----
__global__ __launch_bounds__(256) void ln_pack_k(
    const float* __restrict__ in, const float* __restrict__ g,
    const float* __restrict__ bet, _Float16* __restrict__ out,
    int doLN, int srcNCHW) {
  int t = blockIdx.x * 256 + threadIdx.x;   // token id
  int b = t >> 14, hw = t & 16383;
  float mu = 0.f, rstd = 1.f;
  if (doLN) {
    float s = 0.f, s2 = 0.f;
    for (int c = 0; c < C_; ++c) {
      float x = srcNCHW ? in[(((size_t)b * C_ + c) << 14) + hw] : in[(size_t)t * C_ + c];
      s += x; s2 += x * x;
    }
    mu = s * (1.f / C_);
    float var = s2 * (1.f / C_) - mu * mu;
    rstd = rsqrtf(var + 1e-5f);
  }
  for (int c = 0; c < C_; ++c) {
    float x = srcNCHW ? in[(((size_t)b * C_ + c) << 14) + hw] : in[(size_t)t * C_ + c];
    float v = doLN ? ((x - mu) * rstd * g[c] + bet[c]) : x;
    out[(size_t)t * C_ + c] = (_Float16)v;
  }
}

// ---- generic WMMA GEMM: 128-M tile, 96-N tile, fused epilogues ----
// EPI 0: f16 store (+bias)   1: f16 store (+bias)*scale
// EPI 2: f32 store +bias +residual from NCHW tensor
// EPI 3: f16 store +bias, exact GELU
// EPI 4: f32 NCHW store +bias +residual from token-major f32
template <int KD, int EPI>
__global__ __launch_bounds__(256) void gemm_k(
    const _Float16* __restrict__ A, int lda,
    const _Float16* __restrict__ Bw, int ldb,
    const float* __restrict__ bias,
    void* __restrict__ outp, int ldo,
    const float* __restrict__ res, float scale) {
  const int tid = threadIdx.x;
  const int wave = tid >> 5, lane = tid & 31;
  const int hs = lane >> 4, l16 = lane & 15;
  const int mrow0 = blockIdx.x * 128 + wave * 16;
  const int n0 = blockIdx.y * 96;

  v8f acc[6];
  v8f zero = {};
#pragma unroll
  for (int j = 0; j < 6; ++j) acc[j] = zero;

  const _Float16* arow = A + (size_t)(mrow0 + l16) * lda;
#pragma unroll
  for (int ks = 0; ks < KD / 32; ++ks) {
    const int k0 = ks * 32;
    v16h a = ldA(arow + k0, hs);
    const _Float16* bp = Bw + (size_t)k0 * ldb + n0;
#pragma unroll
    for (int j = 0; j < 6; ++j) {
      v16h b = ldB(bp + j * 16, ldb);
      acc[j] = __builtin_amdgcn_wmma_f32_16x16x32_f16(false, a, false, b,
                                                      (short)0, acc[j], false, false);
    }
  }

  const int mbase = mrow0 + hs * 8;
#pragma unroll
  for (int j = 0; j < 6; ++j) {
    const int col = n0 + j * 16 + l16;
    const float bv = bias[col];
#pragma unroll
    for (int r = 0; r < 8; ++r) {
      const int row = mbase + r;  // token id
      float v = acc[j][r] + bv;
      if (EPI == 0) {
        ((_Float16*)outp)[(size_t)row * ldo + col] = (_Float16)v;
      } else if (EPI == 1) {
        ((_Float16*)outp)[(size_t)row * ldo + col] = (_Float16)(v * scale);
      } else if (EPI == 2) {
        int bb = row >> 14, hw = row & 16383;
        v += res[(((size_t)bb * C_ + col) << 14) + hw];
        ((float*)outp)[(size_t)row * ldo + col] = v;
      } else if (EPI == 3) {
        v = 0.5f * v * (1.f + erff(v * 0.70710678118654752f));
        ((_Float16*)outp)[(size_t)row * ldo + col] = (_Float16)v;
      } else {  // EPI 4
        int bb = row >> 14, hw = row & 16383;
        v += res[(size_t)row * C_ + col];
        ((float*)outp)[(((size_t)bb * C_ + col) << 14) + hw] = v;
      }
    }
  }
}

// ---- overlapping-window attention, one (qtile=64, head, window) per block ----
__global__ __launch_bounds__(256) void attn_k(
    const _Float16* __restrict__ q, const _Float16* __restrict__ kvb,
    const float* __restrict__ bb, _Float16* __restrict__ o) {
  extern __shared__ _Float16 sm[];
  _Float16* kt = sm;                       // K^T [32][KT_LD]
  _Float16* vv = sm + 32 * KT_LD;          // V   [576][32]
  _Float16* sc = vv + NKT * D_;            // S/P [64][SC_LD]
  float* rowsum = (float*)(sc + 64 * SC_LD);  // [64]
  float* red = rowsum + 64;                   // [256] partial reductions

  const int tid = threadIdx.x;
  const int qt = blockIdx.x, head = blockIdx.y, win = blockIdx.z;
  const int b = win >> 6, wy = (win >> 3) & 7, wx = win & 7;

  // stage K^T and V from padded 24x24 window
  for (int t = tid; t < NKT; t += 256) {
    int iy = t / OWSZ, ix = t - iy * OWSZ;
    int hy = wy * WSZ - 4 + iy, hx = wx * WSZ - 4 + ix;
    union { v8h v[4]; _Float16 e[32]; } ku, vu;
    if ((unsigned)hy < 128u && (unsigned)hx < 128u) {
      size_t tok = (size_t)b * 16384 + (size_t)hy * 128 + hx;
      const _Float16* kp = kvb + tok * 384 + head * D_;
      const _Float16* vp = kp + 192;
#pragma unroll
      for (int i = 0; i < 4; ++i) { ku.v[i] = *(const v8h*)(kp + i * 8); vu.v[i] = *(const v8h*)(vp + i * 8); }
    } else {
      v8h zz = {};
#pragma unroll
      for (int i = 0; i < 4; ++i) { ku.v[i] = zz; vu.v[i] = zz; }
    }
#pragma unroll
    for (int i = 0; i < 4; ++i) *(v8h*)(vv + t * D_ + i * 8) = vu.v[i];
#pragma unroll
    for (int d = 0; d < D_; ++d) kt[d * KT_LD + t] = ku.e[d];
  }
  __syncthreads();

  const int wave = tid >> 5, lane = tid & 31, hs = lane >> 4, l16 = lane & 15;
  v8f zero = {};

  // Phase 1: S = Qs @ K^T; two fragments per iteration to hide WMMA->VALU hazard
  {
    int mi = wave >> 1;
    int qrow = qt * 64 + mi * 16 + l16;
    int iy = qrow >> 4, ix = qrow & 15;
    size_t tok = (size_t)b * 16384 + (size_t)(wy * WSZ + iy) * 128 + wx * WSZ + ix;
    const _Float16* qp = q + tok * C_ + head * D_;
    v16h a = ldA(qp, hs);
    int nbase = (wave & 1) * 288;
    int m = mi * 16 + hs * 8;
#pragma unroll 3
    for (int j = 0; j < 18; j += 2) {
      int n0 = nbase + j * 16;
      int n1 = n0 + 16;
      v16h b0 = ldB(kt + n0, KT_LD);
      v16h b1 = ldB(kt + n1, KT_LD);
      v8f s0 = __builtin_amdgcn_wmma_f32_16x16x32_f16(false, a, false, b0,
                                                      (short)0, zero, false, false);
      v8f s1 = __builtin_amdgcn_wmma_f32_16x16x32_f16(false, a, false, b1,
                                                      (short)0, zero, false, false);
      int c0 = n0 + l16, c1 = n1 + l16;
#pragma unroll
      for (int r = 0; r < 8; ++r) {
        sc[(m + r) * SC_LD + c0] = (_Float16)s0[r];
        sc[(m + r) * SC_LD + c1] = (_Float16)s1[r];
      }
    }
  }
  __syncthreads();

  // Phase 2: exact softmax, 4 threads per q row (all 256 threads busy)
  {
    const int r = tid >> 2, p = tid & 3;       // row, partition
    const int qidx = qt * 64 + r;
    const float* bp = bb + ((size_t)head * 256 + qidx) * NKT;
    const int kbeg = p * 144, kend = kbeg + 144;
    float mx = -1e30f;
    for (int k = kbeg; k < kend; ++k) {
      float s = (float)sc[r * SC_LD + k] + bp[k];
      mx = fmaxf(mx, s);
    }
    red[tid] = mx;
    __syncthreads();
    float rmx = fmaxf(fmaxf(red[r * 4 + 0], red[r * 4 + 1]),
                      fmaxf(red[r * 4 + 2], red[r * 4 + 3]));
    float sum = 0.f;
    for (int k = kbeg; k < kend; ++k) {
      float s = (float)sc[r * SC_LD + k] + bp[k];
      float e = __expf(s - rmx);
      sum += e;
      sc[r * SC_LD + k] = (_Float16)e;
    }
    __syncthreads();
    red[tid] = sum;
    __syncthreads();
    if (p == 0)
      rowsum[r] = red[r * 4 + 0] + red[r * 4 + 1] + red[r * 4 + 2] + red[r * 4 + 3];
  }
  __syncthreads();

  // Phase 3: O = P @ V with two accumulators to break the dependency chain
  {
    int mi = wave >> 1, nj = wave & 1;
    v8f acc0 = zero, acc1 = zero;
    const _Float16* aprow = sc + (size_t)(mi * 16 + l16) * SC_LD;
#pragma unroll 3
    for (int ks = 0; ks < 18; ks += 2) {
      int k0 = ks * 32, k1 = k0 + 32;
      v16h a0 = ldA(aprow + k0, hs);
      v16h a1 = ldA(aprow + k1, hs);
      v16h b0 = ldB(vv + (size_t)k0 * D_ + nj * 16, D_);
      v16h b1 = ldB(vv + (size_t)k1 * D_ + nj * 16, D_);
      acc0 = __builtin_amdgcn_wmma_f32_16x16x32_f16(false, a0, false, b0,
                                                    (short)0, acc0, false, false);
      acc1 = __builtin_amdgcn_wmma_f32_16x16x32_f16(false, a1, false, b1,
                                                    (short)0, acc1, false, false);
    }
    v8f acc = acc0 + acc1;
    int m0 = mi * 16 + hs * 8;
    int col = head * D_ + nj * 16 + l16;
#pragma unroll
    for (int r = 0; r < 8; ++r) {
      int m = m0 + r;
      int qidx = qt * 64 + m;
      int iy = qidx >> 4, ix = qidx & 15;
      size_t tok = (size_t)b * 16384 + (size_t)(wy * WSZ + iy) * 128 + wx * WSZ + ix;
      o[tok * C_ + col] = (_Float16)(acc[r] / rowsum[m]);
    }
  }
}

extern "C" void kernel_launch(void* const* d_in, const int* in_sizes, int n_in,
                              void* d_out, int out_size, void* d_ws, size_t ws_size,
                              hipStream_t stream) {
  const float* x       = (const float*)d_in[0];
  const float* depth   = (const float*)d_in[1];
  const int*   rpi     = (const int*)d_in[2];
  const float* n1w     = (const float*)d_in[3];
  const float* n1b     = (const float*)d_in[4];
  const float* qkv_w   = (const float*)d_in[5];
  const float* qkv_b   = (const float*)d_in[6];
  const float* dproj_w = (const float*)d_in[7];
  const float* dproj_b = (const float*)d_in[8];
  const float* rpb     = (const float*)d_in[9];
  const float* proj_w  = (const float*)d_in[10];
  const float* proj_b  = (const float*)d_in[11];
  const float* n2w     = (const float*)d_in[12];
  const float* n2b     = (const float*)d_in[13];
  const float* fc1_w   = (const float*)d_in[14];
  const float* fc1_b   = (const float*)d_in[15];
  const float* fc2_w   = (const float*)d_in[16];
  const float* fc2_b   = (const float*)d_in[17];

  char* ws = (char*)d_ws;
  size_t off = 0;
  auto carve = [&](size_t bytes) -> char* {
    char* p = ws + off;
    off = (off + bytes + 255) & ~(size_t)255;
    return p;
  };
  _Float16* wkv    = (_Float16*)carve(73728 * 2);
  _Float16* dT     = (_Float16*)carve(36864 * 2);
  _Float16* pw     = (_Float16*)carve(36864 * 2);
  _Float16* f1     = (_Float16*)carve(73728 * 2);
  _Float16* f2     = (_Float16*)carve(73728 * 2);
  float*    biasb  = (float*)   carve((size_t)NH_ * 256 * NKT * 4);
  _Float16* xln    = (_Float16*)carve((size_t)NTOK * C_ * 2);
  _Float16* dep16  = (_Float16*)carve((size_t)NTOK * C_ * 2);
  _Float16* kvbuf  = (_Float16*)carve((size_t)NTOK * 384 * 2);
  _Float16* qbuf   = (_Float16*)carve((size_t)NTOK * C_ * 2);
  _Float16* attno  = (_Float16*)carve((size_t)NTOK * C_ * 2);
  float*    out1   = (float*)   carve((size_t)NTOK * C_ * 4);
  _Float16* ln2b   = (_Float16*)carve((size_t)NTOK * C_ * 2);
  _Float16* h1     = (_Float16*)carve((size_t)NTOK * 384 * 2);
  (void)ws_size; (void)in_sizes; (void)n_in; (void)out_size;

  // weight prep + bias gather
  prep_w_k<<<(294912 + 255) / 256, 256, 0, stream>>>(qkv_w, dproj_w, proj_w, fc1_w, fc2_w,
                                                     wkv, dT, pw, f1, f2);
  bias_k<<<(NH_ * 256 * NKT + 255) / 256, 256, 0, stream>>>(rpi, rpb, biasb);

  // LN1 + pack (NCHW->token-major f16); depth pack (no LN)
  ln_pack_k<<<NTOK / 256, 256, 0, stream>>>(x, n1w, n1b, xln, 1, 1);
  ln_pack_k<<<NTOK / 256, 256, 0, stream>>>(depth, n1w, n1b, dep16, 0, 1);

  // KV = LN(x) @ Wkv + bkv      (N=384)
  gemm_k<192, 0><<<dim3(NTOK / 128, 4), 256, 0, stream>>>(xln, 192, wkv, 384,
                                                          qkv_b + 192, kvbuf, 384, nullptr, 1.f);
  // Q = depth @ dproj^T + b, *1/sqrt(32)
  gemm_k<192, 1><<<dim3(NTOK / 128, 2), 256, 0, stream>>>(dep16, 192, dT, 192,
                                                          dproj_b, qbuf, 192, nullptr,
                                                          0.17677669529663687f);
  // attention
  const int smem = (32 * KT_LD + NKT * D_ + 64 * SC_LD) * 2 + 64 * 4 + 256 * 4;
  attn_k<<<dim3(4, NH_, 128), 256, smem, stream>>>(qbuf, kvbuf, biasb, attno);

  // proj + residual(x)
  gemm_k<192, 2><<<dim3(NTOK / 128, 2), 256, 0, stream>>>(attno, 192, pw, 192,
                                                          proj_b, out1, 192, x, 1.f);
  // LN2
  ln_pack_k<<<NTOK / 256, 256, 0, stream>>>(out1, n2w, n2b, ln2b, 1, 0);
  // FC1 + GELU
  gemm_k<192, 3><<<dim3(NTOK / 128, 4), 256, 0, stream>>>(ln2b, 192, f1, 384,
                                                          fc1_b, h1, 384, nullptr, 1.f);
  // FC2 + residual(out1) -> NCHW f32 output
  gemm_k<384, 4><<<dim3(NTOK / 128, 2), 256, 0, stream>>>(h1, 384, f2, 192,
                                                          fc2_b, d_out, 192, out1, 1.f);
}
